// RecurrentGCN_22282290332403
// MI455X (gfx1250) — compile-verified, matrix operand order checked by reading, not run on hardware
//
#include <hip/hip_runtime.h>
#include <hip/hip_bf16.h>

typedef __attribute__((ext_vector_type(16))) _Float16 v16h;
typedef __attribute__((ext_vector_type(8)))  float    v8f;

#define FDIM 32
#define HID  256
#define CIN  288
#define NTILES 16               // 256 cols / 16
#define PACKED_PER_W 8192       // 16 tiles * 32 lanes * 16 halfs

// ---------------------------------------------------------------------------
// Pre-pass: fold Wz[0][:32]+Wz[1][:32] (and Wh likewise) to f16, laid out in
// the exact per-lane VGPR packing the WMMA B-matrix expects:
//   packed[t][lane][j]:  n = t*16 + (lane&15), khalf = lane>>4,
//   j<8  -> k = khalf*8 + j ;  j>=8 -> k = 16 + khalf*8 + (j-8)
// ---------------------------------------------------------------------------
__global__ void pack_weights_kernel(const float* __restrict__ Wz,
                                    const float* __restrict__ Wh,
                                    _Float16* __restrict__ outZ,
                                    _Float16* __restrict__ outH) {
  int p = blockIdx.x * blockDim.x + threadIdx.x;
  if (p >= PACKED_PER_W) return;
  int t = p >> 9;              // 512 packed halfs per column tile
  int l = (p >> 4) & 31;
  int j = p & 15;
  int n = t * 16 + (l & 15);
  int khalf = l >> 4;
  int k = (j < 8) ? (khalf * 8 + j) : (16 + khalf * 8 + (j - 8));
  const int dstride = CIN * HID;           // W is [2, 288, 256] row-major
  float vz = Wz[k * HID + n] + Wz[dstride + k * HID + n];
  float vh = Wh[k * HID + n] + Wh[dstride + k * HID + n];
  outZ[p] = (_Float16)vz;
  outH[p] = (_Float16)vh;
}

// ---------------------------------------------------------------------------
// Main kernel: one wave32 per 16-row tile of x.
//   Z  = sigmoid(x @ WzEff + bz)        (16 WMMAs, K=32 == F)
//   Ht = tanh   (x @ WhEff + bh)        (16 WMMAs)
//   out = relu((1-Z)*Ht) @ Wl + bl      (per-lane partials + xor butterfly)
// Steady-state global traffic = x stream only; weights/biases live in LDS.
// ---------------------------------------------------------------------------
__global__ __launch_bounds__(256) void dcrnn_step_kernel(
    const float* __restrict__ x,
    const _Float16* __restrict__ wzP,
    const _Float16* __restrict__ whP,
    const float* __restrict__ bz,
    const float* __restrict__ bh,
    const float* __restrict__ Wl,
    const float* __restrict__ bl,
    float* __restrict__ out,
    int nrows) {
  __shared__ __align__(32) _Float16 ldsW[2 * PACKED_PER_W];  // 32 KB packed B
  __shared__ float ldsBZ[HID];
  __shared__ float ldsBH[HID];
  __shared__ float ldsWL[HID];

  // Cooperative stage: packed weights (uint4 = 16B) + bias/proj vectors.
  {
    const uint4* srcZ = (const uint4*)wzP;
    const uint4* srcH = (const uint4*)whP;
    uint4* dst = (uint4*)ldsW;
    for (int i = threadIdx.x; i < PACKED_PER_W / 8; i += 256) {
      dst[i] = srcZ[i];
      dst[PACKED_PER_W / 8 + i] = srcH[i];
    }
    int i = threadIdx.x;            // blockDim == HID == 256: one elem each
    ldsBZ[i] = bz[i];
    ldsBH[i] = bh[i];
    ldsWL[i] = Wl[i];
  }
  __syncthreads();

  const int lane  = threadIdx.x & 31;
  const int wave  = threadIdx.x >> 5;
  const int tile  = blockIdx.x * 8 + wave;
  const long r0   = (long)tile * 16;
  if (r0 >= nrows) return;                 // wave-uniform: EXEC stays all-1

  const int m     = lane & 15;             // row in tile (A) / col in tile (B,C)
  const int khalf = lane >> 4;

  // A-matrix 16x32 f16 per ISA layout: lane holds K groups
  // [khalf*8, +8) and [16+khalf*8, +8) of row r0+m.
  const float* xr = x + (r0 + m) * FDIM;
  v16h a;
#pragma unroll
  for (int j = 0; j < 8; ++j) {
    a[j]     = (_Float16)xr[khalf * 8 + j];
    a[8 + j] = (_Float16)xr[16 + khalf * 8 + j];
  }

  const float blv = bl[0];
  float partial[8];
#pragma unroll
  for (int v = 0; v < 8; ++v) partial[v] = 0.f;

  const v16h* ldsZ = (const v16h*)(ldsW);
  const v16h* ldsH = (const v16h*)(ldsW + PACKED_PER_W);

#pragma unroll 2
  for (int t = 0; t < NTILES; ++t) {
    const int n = t * 16 + m;              // global output column for this lane
    v16h bzt = ldsZ[t * 32 + lane];        // ds_load_b128 x2
    v16h bht = ldsH[t * 32 + lane];
    v8f accZ = {};
    v8f accH = {};
    accZ = __builtin_amdgcn_wmma_f32_16x16x32_f16(
        false, a, false, bzt, (short)0, accZ, false, false);
    accH = __builtin_amdgcn_wmma_f32_16x16x32_f16(
        false, a, false, bht, (short)0, accH, false, false);

    const float biasZ = ldsBZ[n];
    const float biasH = ldsBH[n];
    const float wl    = ldsWL[n];
#pragma unroll
    for (int v = 0; v < 8; ++v) {
      // sigmoid via v_exp + v_rcp (fast hw reciprocal, no IEEE div expansion)
      float z  = __builtin_amdgcn_rcpf(1.f + __expf(-(accZ[v] + biasZ)));
      // tanh(y) = (1-e)/(1+e), e = exp(-2y)
      float e  = __expf(-2.f * (accH[v] + biasH));
      float ht = (1.f - e) * __builtin_amdgcn_rcpf(1.f + e);
      float hv = (1.f - z) * ht;
      hv = hv > 0.f ? hv : 0.f;                                // relu
      partial[v] += hv * wl;
    }
  }

  // C layout: VGPR v holds M=v (lanes 0-15) and M=v+8 (lanes 16-31), N=lane&15.
  // Reduce the dot over N within each 16-lane half via xor butterfly.
#pragma unroll
  for (int v = 0; v < 8; ++v) {
    float s = partial[v];
    s += __shfl_xor(s, 1, 32);
    s += __shfl_xor(s, 2, 32);
    s += __shfl_xor(s, 4, 32);
    s += __shfl_xor(s, 8, 32);
    partial[v] = s;
  }
  if (m == 0) {
    const long base = r0 + (long)khalf * 8;
#pragma unroll
    for (int v = 0; v < 8; ++v)
      out[base + v] = partial[v] + blv;
  }
}

// ---------------------------------------------------------------------------
// Inputs (setup_inputs order): x, edge_index, edge_weight, Wz, bz, Wr, br,
// Wh, bh, Wl, bl.  edge_*, Wr, br are mathematically dead (H0=0, K=1).
// ---------------------------------------------------------------------------
extern "C" void kernel_launch(void* const* d_in, const int* in_sizes, int n_in,
                              void* d_out, int out_size, void* d_ws, size_t ws_size,
                              hipStream_t stream) {
  const float* x  = (const float*)d_in[0];
  const float* Wz = (const float*)d_in[3];
  const float* bz = (const float*)d_in[4];
  const float* Wh = (const float*)d_in[7];
  const float* bh = (const float*)d_in[8];
  const float* Wl = (const float*)d_in[9];
  const float* bl = (const float*)d_in[10];
  float* out = (float*)d_out;

  const int nrows = in_sizes[0] / FDIM;

  _Float16* wsZ = (_Float16*)d_ws;
  _Float16* wsH = wsZ + PACKED_PER_W;      // 16 KB + 16 KB of workspace

  pack_weights_kernel<<<(PACKED_PER_W + 255) / 256, 256, 0, stream>>>(
      Wz, Wh, wsZ, wsH);

  const int waves  = (nrows + 15) / 16;    // 6250 wave-tiles
  const int blocks = (waves + 7) / 8;      // 8 waves / block
  dcrnn_step_kernel<<<blocks, 256, 0, stream>>>(
      x, wsZ, wsH, bz, bh, Wl, bl, out, nrows);
}
